// PointNet_7430293422598
// MI455X (gfx1250) — matrix operand order repaired and checked by reading, not compiled, not run on hardware
//
#include <hip/hip_runtime.h>

// ---------------------------------------------------------------------------
// PointNet++ SSG pipeline for MI455X (gfx1250, wave32, WMMA).
// Shared-MLP layers run as bf16 WMMA GEMMs (v_wmma_f32_16x16x32_bf16),
// roofline-optimal precision here (memory-bound: ~45 GFLOP vs ~0.8 GB bf16
// activation traffic -> ~35us at 23.3 TB/s HBM). Weight panels are staged
// into LDS with gfx1250 async-to-LDS DMA (ASYNCcnt) and read as ds_load.
// ---------------------------------------------------------------------------

typedef __attribute__((ext_vector_type(16))) __bf16        v16bf;
typedef __attribute__((ext_vector_type(8)))  float         v8f;
typedef __attribute__((ext_vector_type(8)))  unsigned int  v8u;

union BFPack { v8u u; v16bf bf; };

static __device__ __forceinline__ unsigned short f2bf(float f) {
  unsigned int x = __float_as_uint(f);
  unsigned int r = x + 0x7FFFu + ((x >> 16) & 1u);   // round-to-nearest-even
  return (unsigned short)(r >> 16);
}
static __device__ __forceinline__ float bf2f(unsigned short h) {
  return __uint_as_float(((unsigned int)h) << 16);
}

// ---------------------------------------------------------------------------
// Weight conversion: f32 [Cout][Cin] -> bf16 [Cout][CinPad], zero padded.
// ---------------------------------------------------------------------------
__global__ void k_cvt_w(const float* __restrict__ W, unsigned short* __restrict__ Wb,
                        int Cin, int CinPad, int total) {
  int t = blockIdx.x * blockDim.x + threadIdx.x;
  if (t >= total) return;
  int o = t / CinPad, c = t - o * CinPad;
  float v = (c < Cin) ? W[(size_t)o * Cin + c] : 0.f;
  Wb[t] = f2bf(v);
}

// ---------------------------------------------------------------------------
// Farthest point sampling. One workgroup (1024 thr = 32 waves) per batch.
// dists[] kept resident in LDS (<=80KB of the 320KB WGP budget).
// ---------------------------------------------------------------------------
__global__ void k_fps(const float* __restrict__ xyz, int* __restrict__ inds,
                      int N, int S) {
  extern __shared__ unsigned char smraw[];
  float* dists = (float*)smraw;
  float* rv    = dists + N;
  int*   ri    = (int*)(rv + blockDim.x);
  float* lastp = (float*)(ri + blockDim.x);

  const int b = blockIdx.x, tid = threadIdx.x, T = blockDim.x;
  const float* px = xyz + (size_t)b * N * 3;
  int* out = inds + (size_t)b * S;

  for (int i = tid; i < N; i += T) dists[i] = 1e10f;
  if (tid == 0) {
    out[0] = 0;
    lastp[0] = px[0]; lastp[1] = px[1]; lastp[2] = px[2];
  }
  __syncthreads();

  for (int t = 1; t < S; t++) {
    const float lx = lastp[0], ly = lastp[1], lz = lastp[2];
    float best = -1.f; int bi = 0;
    for (int i = tid; i < N; i += T) {
      float dx = px[3*i+0] - lx, dy = px[3*i+1] - ly, dz = px[3*i+2] - lz;
      float d = dx*dx + dy*dy + dz*dz;
      float m = dists[i]; m = d < m ? d : m; dists[i] = m;
      if (m > best) { best = m; bi = i; }
    }
    rv[tid] = best; ri[tid] = bi;
    __syncthreads();
    for (int s = T >> 1; s > 0; s >>= 1) {
      if (tid < s) {
        float v2 = rv[tid + s];
        if (v2 > rv[tid] || (v2 == rv[tid] && ri[tid + s] < ri[tid])) {
          rv[tid] = v2; ri[tid] = ri[tid + s];
        }
      }
      __syncthreads();
    }
    if (tid == 0) {
      int w = ri[0]; out[t] = w;
      lastp[0] = px[3*w]; lastp[1] = px[3*w+1]; lastp[2] = px[3*w+2];
    }
    __syncthreads();
  }
}

// new_xyz[b][s][d] = xyz[b][inds[b][s]][d]
__global__ void k_gather_xyz(const float* __restrict__ xyz, const int* __restrict__ inds,
                             float* __restrict__ nx, int N, int S, int total) {
  int t = blockIdx.x * blockDim.x + threadIdx.x;
  if (t >= total) return;
  int d = t % 3, rest = t / 3;
  int b = rest / S, s = rest - b * S;
  nx[t] = xyz[((size_t)b * N + inds[(size_t)b * S + s]) * 3 + d];
}

// ---------------------------------------------------------------------------
// Ball query: one wave per query point; wave32 ballot-based ordered
// compaction picks the first K in-radius indices (matches top_k on keys).
// ---------------------------------------------------------------------------
__global__ void k_ballq(const float* __restrict__ xyz, const float* __restrict__ nx,
                        int* __restrict__ idx, int N, int S, int K, float r2, int nQ) {
  const int lane = threadIdx.x & 31;
  const int q = blockIdx.x * (blockDim.x >> 5) + (threadIdx.x >> 5);
  if (q >= nQ) return;                 // wave-uniform
  const int b = q / S;
  const float qx = nx[(size_t)q*3+0], qy = nx[(size_t)q*3+1], qz = nx[(size_t)q*3+2];
  const float* px = xyz + (size_t)b * N * 3;
  int* io = idx + (size_t)q * K;

  int cnt = 0, first = -1;
  for (int base = 0; base < N && cnt < K; base += 32) {
    int i = base + lane;
    bool in = false;
    if (i < N) {
      float dx = px[3*i+0]-qx, dy = px[3*i+1]-qy, dz = px[3*i+2]-qz;
      in = (dx*dx + dy*dy + dz*dz) < r2;
    }
    unsigned m = (unsigned)__ballot(in);
    if (in) {
      int pos = cnt + __popc(m & ((1u << lane) - 1u));
      if (pos < K) io[pos] = i;
    }
    if (first < 0 && m) first = base + __builtin_ctz(m);
    cnt += __popc(m);
  }
  if (first < 0) first = 0;
  for (int j = cnt + lane; j < K; j += 32) io[j] = first;
}

// ---------------------------------------------------------------------------
// Build grouped bf16 activation matrix X[Cpad][P], P = B*S*K.
// ---------------------------------------------------------------------------
__global__ void k_buildx_sa(const float* __restrict__ xyz, const float* __restrict__ nx,
                            const int* __restrict__ idx, const float* __restrict__ feats,
                            unsigned short* __restrict__ X,
                            int N, int S, int K, int Cfeat, int P, float invR, int B) {
  int c = blockIdx.y;
  int p = blockIdx.x * blockDim.x + threadIdx.x;
  if (p >= P) return;
  int rest = p / K;                    // b*S + s
  int b = rest / S;
  float v = 0.f;
  if (c < 3) {
    int i = idx[p];
    v = (xyz[((size_t)b * N + i) * 3 + c] - nx[(size_t)rest * 3 + c]) * invR;
  } else if (c < 3 + Cfeat) {
    int i = idx[p];
    v = feats[((size_t)(c - 3) * B + b) * N + i];
  }
  X[(size_t)c * P + p] = f2bf(v);
}

// ---------------------------------------------------------------------------
// WMMA GEMM + BN(scale,bias) + ReLU.  Y[o,p] = relu(g[o]*sum_c W[o,c]X[c,p]+b[o])
//
// Block = 4 waves, all sharing one o-group of 64 output channels. The 64 x
// CinPad bf16 weight panel is DMA'd to LDS with global_load_async_to_lds_b128
// (ASYNCcnt, drained by s_wait_asynccnt) and read back as ds_load. Each wave
// owns one 16-column p-tile and keeps 4 f32 accumulators (16x64 tile), so one
// B fragment feeds 4 v_wmma_f32_16x16x32_bf16 per K-step.
// Fragment K order: K = 16*(lane/16) + element for both A and B.
// ---------------------------------------------------------------------------
__global__ void k_gemm_bn_relu(const unsigned short* __restrict__ Wb,
                               const unsigned short* __restrict__ X,
                               const float* __restrict__ g, const float* __restrict__ beta,
                               unsigned short* __restrict__ Y, float* __restrict__ Yf,
                               int CinPad, int Cout, int P) {
  extern __shared__ unsigned char smraw[];
  unsigned short* sA = (unsigned short*)smraw;

  const int oGroups = Cout >> 6;                     // 64 rows per group
  const int oGroup  = blockIdx.x % oGroups;
  const int pBlock  = blockIdx.x / oGroups;          // p-major: X panel stays in L2
  const int waveId  = threadIdx.x >> 5;
  const int lane    = threadIdx.x & 31;

  // ---- async DMA of the 64 x CinPad weight panel into LDS ----
  {
    const size_t panelBytes = (size_t)64 * CinPad * 2;      // multiple of 4096
    const unsigned char* gsrc =
        (const unsigned char*)(Wb + (size_t)oGroup * 64 * CinPad);
    for (size_t ofs = (size_t)threadIdx.x * 16; ofs < panelBytes;
         ofs += (size_t)blockDim.x * 16) {
      unsigned lds = (unsigned)(size_t)(smraw + ofs);
      unsigned long long ga = (unsigned long long)(size_t)(gsrc + ofs);
      asm volatile("global_load_async_to_lds_b128 %0, %1, off"
                   :: "v"(lds), "v"(ga) : "memory");
    }
    asm volatile("s_wait_asynccnt 0" ::: "memory");
    __syncthreads();
  }

  const int pTile = pBlock * 4 + waveId;
  const int m  = lane & 15;
  const bool hi = lane >= 16;
  const int pcol = (pTile << 4) + m;                 // column index (N = lane&15)
  const int aoff = hi ? 16 : 0;                      // A: K-offset per half-wave
  const int boff = hi ? 16 : 0;                      // B: K-offset per half-wave

  v8f acc0 = {}, acc1 = {}, acc2 = {}, acc3 = {};

  for (int kb = 0; kb < CinPad; kb += 32) {
    if (kb + 32 < CinPad) {                          // gfx1250 global_prefetch_b8
      __builtin_prefetch(X + (size_t)(kb + 32 + boff) * P + pcol, 0, 1);
    }
    // B fragment: column pcol; K = kb + boff + {0..15}, packed 2/dword
    BFPack bm;
    const unsigned short* xc = X + (size_t)(kb + boff) * P + pcol;
#pragma unroll
    for (int j = 0; j < 8; j++) {
      unsigned int lo  = xc[(size_t)(2 * j) * P];
      unsigned int hiw = xc[(size_t)(2 * j + 1) * P];
      bm.u[j] = lo | (hiw << 16);
    }
    // 4 A fragments from LDS (rows t*16+m), dwords {0..7}->K{0,2,..,14}
#pragma unroll
    for (int t = 0; t < 4; t++) {
      BFPack a;
      const unsigned int* ar =
          (const unsigned int*)(sA + (size_t)(t * 16 + m) * CinPad + kb + aoff);
      a.u[0] = ar[0];  a.u[1] = ar[1];  a.u[2] = ar[2];  a.u[3] = ar[3];
      a.u[4] = ar[4];  a.u[5] = ar[5];  a.u[6] = ar[6];  a.u[7] = ar[7];
      v8f& acc = (t == 0) ? acc0 : (t == 1) ? acc1 : (t == 2) ? acc2 : acc3;
      acc = __builtin_amdgcn_wmma_f32_16x16x32_bf16(
          false, a.bf, false, bm.bf, (short)0, acc, false, false);
    }
  }

  // ---- epilogue: BN + ReLU, bf16 (+optional f32) stores ----
#pragma unroll
  for (int t = 0; t < 4; t++) {
    v8f acc = (t == 0) ? acc0 : (t == 1) ? acc1 : (t == 2) ? acc2 : acc3;
    const int obase = (oGroup << 6) + (t << 4) + (hi ? 8 : 0);  // VGPR j -> M=j(+8 hi)
#pragma unroll
    for (int j = 0; j < 8; j++) {
      int o = obase + j;
      float y = g[o] * acc[j] + beta[o];
      y = y > 0.f ? y : 0.f;
      size_t off = (size_t)o * P + pcol;
      Y[off] = f2bf(y);
      if (Yf) Yf[off] = y;
    }
  }
}

// Max-pool over the K neighbor axis: pooled[c][b*S+s] = max_k Y[c][(b*S+s)*K+k]
__global__ void k_maxpool(const unsigned short* __restrict__ Y, float* __restrict__ pooled,
                          int K, int P, int total) {
  int t = blockIdx.x * blockDim.x + threadIdx.x;
  if (t >= total) return;
  int c = blockIdx.y;
  const unsigned short* base = Y + (size_t)c * P + (size_t)t * K;
  float v = bf2f(base[0]);
  for (int k = 1; k < K; k++) { float u = bf2f(base[k]); v = u > v ? u : v; }
  pooled[(size_t)c * total + t] = v;
}

// 3-NN (squared distances, first-index ties) + normalized inverse-distance weights
__global__ void k_threenn(const float* __restrict__ unk, const float* __restrict__ kn,
                          int* __restrict__ i3, float* __restrict__ w3,
                          int nU, int nK, int total) {
  int t = blockIdx.x * blockDim.x + threadIdx.x;
  if (t >= total) return;
  int b = t / nU, i = t - b * nU;
  const float* u = unk + ((size_t)b * nU + i) * 3;
  float bd0 = 1e30f, bd1 = 1e30f, bd2 = 1e30f;
  int bi0 = 0, bi1 = 0, bi2 = 0;
  for (int j = 0; j < nK; j++) {
    const float* kp = kn + ((size_t)b * nK + j) * 3;
    float dx = u[0]-kp[0], dy = u[1]-kp[1], dz = u[2]-kp[2];
    float d = dx*dx + dy*dy + dz*dz;
    if (d < bd0)      { bd2=bd1; bi2=bi1; bd1=bd0; bi1=bi0; bd0=d; bi0=j; }
    else if (d < bd1) { bd2=bd1; bi2=bi1; bd1=d;  bi1=j; }
    else if (d < bd2) { bd2=d;  bi2=j; }
  }
  float w0 = 1.f/(bd0+1e-8f), w1 = 1.f/(bd1+1e-8f), w2 = 1.f/(bd2+1e-8f);
  float s = w0 + w1 + w2;
  i3[t*3+0]=bi0; i3[t*3+1]=bi1; i3[t*3+2]=bi2;
  w3[t*3+0]=w0/s; w3[t*3+1]=w1/s; w3[t*3+2]=w2/s;
}

// FP build-X: rows [0,Ckn) = interpolated known feats, [Ckn,Ckn+Cunk) = skip feats
__global__ void k_buildx_fp(const float* __restrict__ knF, const float* __restrict__ unkF,
                            const int* __restrict__ i3, const float* __restrict__ w3,
                            unsigned short* __restrict__ X,
                            int Ckn, int Cunk, int nU, int nK, int P, int B) {
  int c = blockIdx.y;
  int p = blockIdx.x * blockDim.x + threadIdx.x;
  if (p >= P) return;
  int b = p / nU;
  float v = 0.f;
  if (c < Ckn) {
    const int* ii = i3 + (size_t)p * 3;
    const float* ww = w3 + (size_t)p * 3;
    size_t rb = ((size_t)c * B + b) * nK;
    v = ww[0]*knF[rb+ii[0]] + ww[1]*knF[rb+ii[1]] + ww[2]*knF[rb+ii[2]];
  } else if (c < Ckn + Cunk) {
    v = unkF[(size_t)(c - Ckn) * P + p];
  }
  X[(size_t)c * P + p] = f2bf(v);
}

// out[b][c] = max_i F[c][b*n+i]
__global__ void k_final_max(const float* __restrict__ F, float* __restrict__ out,
                            int C, int n, int P, int total) {
  int t = blockIdx.x * blockDim.x + threadIdx.x;
  if (t >= total) return;
  int b = t / C, c = t - b * C;
  const float* base = F + (size_t)c * P + (size_t)b * n;
  float v = base[0];
  for (int i = 1; i < n; i++) { float u = base[i]; v = u > v ? u : v; }
  out[t] = v;
}

// ---------------------------------------------------------------------------
// Host orchestration
// ---------------------------------------------------------------------------
struct LayerCfg { int co, ci, cp; };
static const LayerCfg LCFG[16] = {
  {64,3,32},   {64,64,64},   {128,64,64},     // SA1
  {128,131,160},{128,128,128},{256,128,128},  // SA2
  {128,259,288},{128,128,128},{256,128,128},  // SA3
  {128,259,288},{128,128,128},{256,128,128},  // SA4
  {512,512,512},{512,512,512},                // FP1
  {512,768,768},{512,512,512}                 // FP2
};

extern "C" void kernel_launch(void* const* d_in, const int* in_sizes, int n_in,
                              void* d_out, int out_size, void* d_ws, size_t ws_size,
                              hipStream_t stream) {
  (void)in_sizes; (void)n_in; (void)out_size; (void)ws_size;
  const int B = 4;
  const int Ns[4] = {20000, 2048, 1024, 512};
  const int Ss[4] = {2048, 1024, 512, 256};
  const int Ks[4] = {64, 32, 16, 16};
  const float Rs[4] = {0.2f, 0.4f, 0.8f, 1.2f};
  const int Cf[4] = {0, 128, 256, 256};

  const float* pc = (const float*)d_in[0];
  const float *Wp[16], *Gp[16], *Bp[16];
  { int q = 1;
    for (int l = 0; l < 16; l++) { Wp[l]=(const float*)d_in[q++]; Gp[l]=(const float*)d_in[q++]; Bp[l]=(const float*)d_in[q++]; } }

  unsigned char* ws = (unsigned char*)d_ws;
  size_t off = 0;
  auto take = [&](size_t bytes) { off = (off + 255) & ~(size_t)255; size_t o = off; off += bytes; return o; };

  size_t oWb[16];
  for (int l = 0; l < 16; l++) oWb[l] = take((size_t)LCFG[l].co * LCFG[l].cp * 2);
  size_t oXa = take((size_t)64  * 524288 * 2);   // 64 MB ping
  size_t oXb = take((size_t)128 * 524288 * 2);   // 128 MB pong
  size_t oInds[4], oNx[4], oIdx[4];
  for (int st = 0; st < 4; st++) {
    oInds[st] = take((size_t)B * Ss[st] * 4);
    oNx[st]   = take((size_t)B * Ss[st] * 3 * 4);
    oIdx[st]  = take((size_t)B * Ss[st] * Ks[st] * 4);
  }
  size_t oF1  = take((size_t)128 * B * 2048 * 4);
  size_t oF2  = take((size_t)256 * B * 1024 * 4);
  size_t oF3  = take((size_t)256 * B * 512  * 4);
  size_t oF4  = take((size_t)256 * B * 256  * 4);
  size_t oFp1 = take((size_t)512 * B * 512  * 4);
  size_t oFp2 = take((size_t)512 * B * 1024 * 4);
  size_t oI3a = take((size_t)B * 512 * 3 * 4),  oW3a = take((size_t)B * 512 * 3 * 4);
  size_t oI3b = take((size_t)B * 1024 * 3 * 4), oW3b = take((size_t)B * 1024 * 3 * 4);
  size_t oPool[4] = {oF1, oF2, oF3, oF4};

  unsigned short* Xa = (unsigned short*)(ws + oXa);
  unsigned short* Xb = (unsigned short*)(ws + oXb);

  // Convert all weights to padded bf16 once.
  for (int l = 0; l < 16; l++) {
    int total = LCFG[l].co * LCFG[l].cp;
    k_cvt_w<<<(total + 255) / 256, 256, 0, stream>>>(
        Wp[l], (unsigned short*)(ws + oWb[l]), LCFG[l].ci, LCFG[l].cp, total);
  }

  auto gemm = [&](int l, const unsigned short* Xin, unsigned short* Yout, float* Yf, int P) {
    int blocks = (LCFG[l].co >> 6) * (P >> 6);       // o-groups of 64 x p-blocks of 64
    size_t smem = (size_t)64 * LCFG[l].cp * 2;       // weight panel in LDS
    k_gemm_bn_relu<<<blocks, 128, smem, stream>>>(
        (const unsigned short*)(ws + oWb[l]), Xin, Gp[l], Bp[l], Yout, Yf,
        LCFG[l].cp, LCFG[l].co, P);
  };

  // ---- SA stages ----
  for (int st = 0; st < 4; st++) {
    const int N = Ns[st], S = Ss[st], K = Ks[st], P = B * S * K;
    const float* xyzIn = (st == 0) ? pc : (const float*)(ws + oNx[st - 1]);
    float* nx  = (float*)(ws + oNx[st]);
    int* inds  = (int*)(ws + oInds[st]);
    int* bidx  = (int*)(ws + oIdx[st]);
    const float* feats = (st == 0) ? nullptr : (const float*)(ws + oPool[st - 1]);

    size_t smem = (size_t)N * 4 + 1024 * 8 + 16;
    k_fps<<<B, 1024, smem, stream>>>(xyzIn, inds, N, S);
    k_gather_xyz<<<(B * S * 3 + 255) / 256, 256, 0, stream>>>(xyzIn, inds, nx, N, S, B * S * 3);
    k_ballq<<<(B * S + 7) / 8, 256, 0, stream>>>(xyzIn, nx, bidx, N, S, K, Rs[st] * Rs[st], B * S);

    int l0 = st * 3;
    k_buildx_sa<<<dim3((P + 255) / 256, LCFG[l0].cp), 256, 0, stream>>>(
        xyzIn, nx, bidx, feats, Xa, N, S, K, Cf[st], P, 1.f / Rs[st], B);

    gemm(l0 + 0, Xa, Xb, nullptr, P);
    gemm(l0 + 1, Xb, Xa, nullptr, P);
    gemm(l0 + 2, Xa, Xb, nullptr, P);

    k_maxpool<<<dim3((B * S + 255) / 256, LCFG[l0 + 2].co), 256, 0, stream>>>(
        Xb, (float*)(ws + oPool[st]), K, P, B * S);
  }

  // ---- FP1: unknown = xyz3(512), known = xyz4(256); feats f3 + interp(f4) ----
  {
    const float* nx3 = (const float*)(ws + oNx[2]);
    const float* nx4 = (const float*)(ws + oNx[3]);
    int P = B * 512;
    k_threenn<<<(P + 255) / 256, 256, 0, stream>>>(nx3, nx4, (int*)(ws + oI3a),
                                                   (float*)(ws + oW3a), 512, 256, P);
    k_buildx_fp<<<dim3(P / 256, LCFG[12].cp), 256, 0, stream>>>(
        (const float*)(ws + oF4), (const float*)(ws + oF3),
        (const int*)(ws + oI3a), (const float*)(ws + oW3a),
        Xa, 256, 256, 512, 256, P, B);
    gemm(12, Xa, Xb, nullptr, P);
    gemm(13, Xb, Xa, (float*)(ws + oFp1), P);
  }

  // ---- FP2: unknown = xyz2(1024), known = xyz3(512); feats f2 + interp(fp1) ----
  {
    const float* nx2 = (const float*)(ws + oNx[1]);
    const float* nx3 = (const float*)(ws + oNx[2]);
    int P = B * 1024;
    k_threenn<<<(P + 255) / 256, 256, 0, stream>>>(nx2, nx3, (int*)(ws + oI3b),
                                                   (float*)(ws + oW3b), 1024, 512, P);
    k_buildx_fp<<<dim3(P / 256, LCFG[14].cp), 256, 0, stream>>>(
        (const float*)(ws + oFp1), (const float*)(ws + oF2),
        (const int*)(ws + oI3b), (const float*)(ws + oW3b),
        Xa, 512, 256, 1024, 512, P, B);
    gemm(14, Xa, Xb, nullptr, P);
    gemm(15, Xb, Xa, (float*)(ws + oFp2), P);

    k_final_max<<<(B * 512 + 255) / 256, 256, 0, stream>>>(
        (const float*)(ws + oFp2), (float*)d_out, 512, 1024, P, B * 512);
  }
}
